// Attention_20564303413254
// MI455X (gfx1250) — matrix operand order repaired and checked by reading, not compiled, not run on hardware
//
#include <hip/hip_runtime.h>
#include <hip/hip_bf16.h>

// ---------------------------------------------------------------------------
// Shapes (fixed by the reference)
// ---------------------------------------------------------------------------
#define SEQ 2048
#define DMODEL 4096
#define HQ 32
#define HK 8
#define HD 128
#define GQA (HQ / HK)   // 4

typedef __attribute__((ext_vector_type(16))) _Float16 v16h;
typedef __attribute__((ext_vector_type(8)))  float    v8f;

// ---------------------------------------------------------------------------
// WMMA fragment loaders (CDNA5 wave32 layouts, CDNA5 ISA §7.12.2)
// ---------------------------------------------------------------------------
__device__ __forceinline__ v16h load_frag_a(const _Float16* base, int ld) {
  const int lane = threadIdx.x & 31;
  const int hg = lane >> 4;
  const int r  = lane & 15;
  const _Float16* p = base + (size_t)r * ld + hg * 8;
  union { v16h v; uint4 u[2]; } f;
  f.u[0] = *(const uint4*)(p);
  f.u[1] = *(const uint4*)(p + 16);
  return f.v;
}

__device__ __forceinline__ v16h load_frag_b(const _Float16* baseT, int ld) {
  const int lane = threadIdx.x & 31;
  const int hg = lane >> 4;
  const int n  = lane & 15;
  const _Float16* p = baseT + (size_t)n * ld + hg * 8;
  union { v16h v; uint4 u[2]; } f;
  f.u[0] = *(const uint4*)(p);
  f.u[1] = *(const uint4*)(p + 16);
  return f.v;
}

__device__ __forceinline__ v8f wmma_f16(v16h a, v16h b, v8f c) {
  return __builtin_amdgcn_wmma_f32_16x16x32_f16(false, a, false, b, (short)0, c,
                                                false, false);
}

// Async global -> LDS copy (CDNA5 VGLOBAL op 98, tracked by ASYNCcnt).
// lds_off is the wave-relative LDS byte address (VDST), gptr the 64-bit VA.
__device__ __forceinline__ void async_load_b128(unsigned lds_off,
                                                const void* gptr) {
  asm volatile("global_load_async_to_lds_b128 %0, %1, off"
               :: "v"(lds_off), "v"(gptr)
               : "memory");
}

__device__ __forceinline__ void wait_asynccnt0() {
  asm volatile("s_wait_asynccnt 0x0" ::: "memory");
}

// ---------------------------------------------------------------------------
// Elementwise conversion / packing kernels
// ---------------------------------------------------------------------------
__global__ void cvt_f32_to_f16(const float* __restrict__ in,
                               _Float16* __restrict__ out, int n) {
  int i = blockIdx.x * blockDim.x + threadIdx.x;
  if (i < n) out[i] = (_Float16)in[i];
}

// W [K][N] fp32  ->  Wt [N][K] f16
__global__ void transcvt(const float* __restrict__ W, _Float16* __restrict__ Wt,
                         int K, int N) {
  int i = blockIdx.x * blockDim.x + threadIdx.x;
  if (i >= K * N) return;
  int k = i / N;
  int n = i - k * N;
  Wt[(size_t)n * K + k] = (_Float16)W[i];
}

// raw [S][nheads*128] fp32 -> out [nheads][S][128] f16 with RoPE (+ scale)
__global__ void rope_pack(const float* __restrict__ raw,
                          const float* __restrict__ cosT,
                          const float* __restrict__ sinT,
                          _Float16* __restrict__ out, int nheads, float scale) {
  int i = blockIdx.x * blockDim.x + threadIdx.x;  // s*nheads*64 + h*64 + j
  if (i >= nheads * SEQ * 64) return;
  int j = i & 63;
  int h = (i >> 6) % nheads;
  int s = i / (nheads * 64);
  size_t src = ((size_t)s * nheads + h) * HD + 2 * j;
  float t0 = raw[src];
  float t1 = raw[src + 1];
  float c  = cosT[s * 64 + j];
  float sn = sinT[s * 64 + j];
  size_t o = (((size_t)h * SEQ) + s) * HD + 2 * j;
  out[o]     = (_Float16)((t0 * c - t1 * sn) * scale);
  out[o + 1] = (_Float16)((t0 * sn + t1 * c) * scale);
}

// raw [S][HK*128] fp32 -> vt [HK][128][S] f16 (transposed per head)
__global__ void transpose_v(const float* __restrict__ raw,
                            _Float16* __restrict__ vt) {
  int i = blockIdx.x * blockDim.x + threadIdx.x;
  if (i >= SEQ * HK * HD) return;
  int s = i >> 10;          // / (HK*HD)
  int c = i & 1023;
  int h = c >> 7;
  int d = c & 127;
  vt[((size_t)h * HD + d) * SEQ + s] = (_Float16)raw[i];
}

// ---------------------------------------------------------------------------
// Async-staged, double-buffered f16 WMMA GEMM (K compile-time):
//   C[M][N] (f32) = A[M][K] (f16, row-major) * Bt[N][K]^T
//
// Block = 8 waves, 128x128 C tile; K-step 64.
//  * 256 threads stage the next A(128x64) + B(128x64) slabs into LDS with
//    GLOBAL_LOAD_ASYNC_TO_LDS_B128 (DMA overlaps with compute, ASYNCcnt).
//  * LDS rows padded to 72 halfs (144 B) -> fragment ds_load_b128 reads are
//    bank-conflict-free (r*36 mod 64 distinct for all 16 rows).
//  * Each wave computes a 64x32 tile: per 32-wide chunk, 6 LDS fragment
//    loads feed 8 independent WMMAs (16 WMMAs per K-step).
// ---------------------------------------------------------------------------
#define LDA_P 72                          // padded LDS row stride (halfs)
#define ATILE_B (128 * LDA_P * 2)         // 18432 bytes per A/B buffer
#define SMEM_TOTAL (4 * ATILE_B)          // A0,A1,B0,B1 = 73728 bytes

template <int K>
__global__ void __launch_bounds__(256) gemm_wmma(const _Float16* __restrict__ A,
                                                 const _Float16* __restrict__ Bt,
                                                 float* __restrict__ C,
                                                 int M, int N) {
  extern __shared__ __align__(16) char smem[];
  const unsigned offA[2] = {0u, (unsigned)ATILE_B};
  const unsigned offB[2] = {2u * ATILE_B, 3u * ATILE_B};

  const int wave = threadIdx.x >> 5;
  const int nbn  = N >> 7;                 // block tiles in N (128 wide)
  const int bm   = blockIdx.x / nbn;
  const int bn   = blockIdx.x - bm * nbn;
  if (bm >= (M >> 7)) return;
  const int wm = wave >> 2;                // 0..1  (64 rows each)
  const int wn = wave & 3;                 // 0..3  (32 cols each)

  const _Float16* Ablk = A + (size_t)(bm * 128) * K;
  const _Float16* Bblk = Bt + (size_t)(bn * 128) * K;

  const int lane = threadIdx.x & 31;
  const int hg = lane >> 4;
  const int r  = lane & 15;

  // per-thread staging slice: row = tid/2 (0..127), 32-half chunk = tid&1
  const int srow = threadIdx.x >> 1;
  const int sch  = (threadIdx.x & 1) * 32;

  auto stage = [&](int buf, int kk) {
    const _Float16* ga = Ablk + (size_t)srow * K + kk + sch;
    const _Float16* gb = Bblk + (size_t)srow * K + kk + sch;
    const unsigned la = offA[buf] + srow * (LDA_P * 2) + sch * 2;
    const unsigned lb = offB[buf] + srow * (LDA_P * 2) + sch * 2;
#pragma unroll
    for (int i = 0; i < 4; ++i) {
      async_load_b128(la + i * 16, ga + i * 8);
      async_load_b128(lb + i * 16, gb + i * 8);
    }
  };

  v8f acc[4][2] = {};

  auto compute = [&](int buf) {
    const _Float16* sa = (const _Float16*)(smem + offA[buf]) + wm * 64 * LDA_P;
    const _Float16* sb = (const _Float16*)(smem + offB[buf]) + wn * 32 * LDA_P;
#pragma unroll
    for (int c = 0; c < 2; ++c) {          // two 32-wide K chunks
      v16h a[4], b[2];
#pragma unroll
      for (int i = 0; i < 4; ++i)
        a[i] = load_frag_a(sa + (16 * i) * LDA_P + c * 32, LDA_P);
#pragma unroll
      for (int j = 0; j < 2; ++j)
        b[j] = load_frag_b(sb + (16 * j) * LDA_P + c * 32, LDA_P);
#pragma unroll
      for (int i = 0; i < 4; ++i)
#pragma unroll
        for (int j = 0; j < 2; ++j)
          acc[i][j] = wmma_f16(a[i], b[j], acc[i][j]);
    }
  };

  // software pipeline: DMA buf^1 while computing buf
  stage(0, 0);
  wait_asynccnt0();
  __syncthreads();

  int buf = 0;
  for (int kk = 0; kk < K; kk += 64) {
    if (kk + 64 < K) stage(buf ^ 1, kk + 64);
    compute(buf);
    wait_asynccnt0();
    __syncthreads();
    buf ^= 1;
  }

  const int m0 = bm * 128 + wm * 64;
  const int n0 = bn * 128 + wn * 32;
#pragma unroll
  for (int i = 0; i < 4; ++i)
#pragma unroll
    for (int j = 0; j < 2; ++j) {
      float* Cb = C + (size_t)(m0 + 16 * i + 8 * hg) * N + n0 + 16 * j + r;
#pragma unroll
      for (int v = 0; v < 8; ++v) Cb[(size_t)v * N] = acc[i][j][v];
    }
}

// ---------------------------------------------------------------------------
// Flash attention (causal, GQA). One wave per (head, 16-row query tile).
//   Qc [HQ][S][128] f16 (pre-scaled by 1/sqrt(HD))
//   Kc [HK][S][128] f16
//   Vt [HK][128][S] f16 (transposed so B-fragments load contiguous key pairs)
//   O  [S][HQ*HD]   f16 (activation for the output projection)
// ---------------------------------------------------------------------------
__global__ void __launch_bounds__(256) attn_fwd(const _Float16* __restrict__ Qc,
                                                const _Float16* __restrict__ Kc,
                                                const _Float16* __restrict__ Vt,
                                                _Float16* __restrict__ O) {
  __shared__ __align__(16) _Float16 pbuf[8][16 * 32];  // per-wave P tile

  const int wid = (blockIdx.x * blockDim.x + threadIdx.x) >> 5;
  const int h   = wid >> 7;    // 0..31 (128 q-tiles per head)
  const int qt  = wid & 127;
  const int kh  = h >> 2;      // GQA: 4 query heads share one KV head

  const int lane = threadIdx.x & 31;
  const int hg = lane >> 4, lid = lane & 15;

  const _Float16* Q  = Qc + (((size_t)h * SEQ) + qt * 16) * HD;
  const _Float16* Kb = Kc + (size_t)kh * SEQ * HD;
  const _Float16* Vb = Vt + (size_t)kh * HD * SEQ;
  _Float16* P = pbuf[threadIdx.x >> 5];

  v16h qa[4];
#pragma unroll
  for (int d = 0; d < 4; ++d) qa[d] = load_frag_a(Q + d * 32, HD);

  v8f acc[8] = {};
  float m[8], l[8];
#pragma unroll
  for (int v = 0; v < 8; ++v) { m[v] = -3.0e38f; l[v] = 0.0f; }

  const int qbase = qt * 16 + 8 * hg;            // query row of acc slot v is qbase+v
  const int nkb = (qt * 16 + 47) >> 5;           // 32-key blocks covering keys <= qmax

  for (int kb = 0; kb < nkb; ++kb) {
    const int key0 = kb * 32;
    const _Float16* K1 = Kb + (size_t)key0 * HD;
    const _Float16* K2 = K1 + 16 * HD;

    v8f s1 = {}, s2 = {};
#pragma unroll
    for (int d = 0; d < 4; ++d) {
      s1 = wmma_f16(qa[d], load_frag_b(K1 + d * 32, HD), s1);
      s2 = wmma_f16(qa[d], load_frag_b(K2 + d * 32, HD), s2);
    }

    float corr[8];
#pragma unroll
    for (int v = 0; v < 8; ++v) {
      const int qrow = qbase + v;
      // causal mask via selects (keeps EXEC all-ones)
      float a = (key0 + lid      <= qrow) ? s1[v] : -3.0e38f;
      float b = (key0 + 16 + lid <= qrow) ? s2[v] : -3.0e38f;
      // row max across the 16 lanes holding this row (xor 1..8 stays in-half)
      float x = fmaxf(a, b);
      x = fmaxf(x, __shfl_xor(x, 1, 32));
      x = fmaxf(x, __shfl_xor(x, 2, 32));
      x = fmaxf(x, __shfl_xor(x, 4, 32));
      x = fmaxf(x, __shfl_xor(x, 8, 32));
      const float mnew = fmaxf(m[v], x);
      corr[v] = __expf(m[v] - mnew);
      const float p1 = __expf(a - mnew);
      const float p2 = __expf(b - mnew);
      float rs = p1 + p2;
      rs += __shfl_xor(rs, 1, 32);
      rs += __shfl_xor(rs, 2, 32);
      rs += __shfl_xor(rs, 4, 32);
      rs += __shfl_xor(rs, 8, 32);
      l[v] = l[v] * corr[v] + rs;
      m[v] = mnew;
      // spill P (C-layout) to LDS in plain [row][col] f16
      const int row = v + 8 * hg;
      P[row * 32 + lid]      = (_Float16)p1;
      P[row * 32 + 16 + lid] = (_Float16)p2;
    }

    // rescale running output accumulators
#pragma unroll
    for (int t = 0; t < 8; ++t) {
      v8f a = acc[t];
#pragma unroll
      for (int v = 0; v < 8; ++v) a[v] *= corr[v];
      acc[t] = a;
    }

    // ensure the DS stores of P have landed before re-reading as an A-fragment
    asm volatile("s_wait_dscnt 0" ::: "memory");
    v16h pf = load_frag_a(P, 32);

#pragma unroll
    for (int t = 0; t < 8; ++t) {
      v16h vf = load_frag_b(Vb + (size_t)(16 * t) * SEQ + key0, SEQ);
      acc[t] = wmma_f16(pf, vf, acc[t]);
    }
  }

  // normalize and emit f16 activation [S][HQ*HD]
  _Float16* Ob = O + (size_t)(qt * 16 + 8 * hg) * (HQ * HD) + h * HD + lid;
#pragma unroll
  for (int v = 0; v < 8; ++v) {
    const float inv = 1.0f / l[v];
#pragma unroll
    for (int t = 0; t < 8; ++t)
      Ob[(size_t)v * (HQ * HD) + t * 16] = (_Float16)(acc[t][v] * inv);
  }
}

// ---------------------------------------------------------------------------
// Host-side launch
// ---------------------------------------------------------------------------
extern "C" void kernel_launch(void* const* d_in, const int* in_sizes, int n_in,
                              void* d_out, int out_size, void* d_ws, size_t ws_size,
                              hipStream_t stream) {
  (void)in_sizes; (void)n_in; (void)out_size; (void)ws_size;

  const float* x  = (const float*)d_in[0];
  const float* fc = (const float*)d_in[1];
  const float* fs = (const float*)d_in[2];
  const float* wq = (const float*)d_in[3];
  const float* wk = (const float*)d_in[4];
  const float* wv = (const float*)d_in[5];
  const float* wo = (const float*)d_in[6];
  float* out = (float*)d_out;

  char* ws = (char*)d_ws;
  size_t off = 0;
  auto take = [&](size_t bytes) -> char* {
    char* p = ws + off;
    off = (off + bytes + 255) & ~(size_t)255;
    return p;
  };

  _Float16* xb   = (_Float16*)take((size_t)SEQ * DMODEL * 2);        // x in f16
  _Float16* wqT  = (_Float16*)take((size_t)DMODEL * DMODEL * 2);     // [4096][4096]
  _Float16* wkT  = (_Float16*)take((size_t)(HK * HD) * DMODEL * 2);  // [1024][4096]
  _Float16* wvT  = (_Float16*)take((size_t)(HK * HD) * DMODEL * 2);
  _Float16* woT  = (_Float16*)take((size_t)DMODEL * DMODEL * 2);
  float*    Qraw = (float*)take((size_t)SEQ * DMODEL * 4);
  float*    Kraw = (float*)take((size_t)SEQ * HK * HD * 4);
  float*    Vraw = (float*)take((size_t)SEQ * HK * HD * 4);
  _Float16* Qc   = (_Float16*)take((size_t)HQ * SEQ * HD * 2);
  _Float16* Kc   = (_Float16*)take((size_t)HK * SEQ * HD * 2);
  _Float16* Vt   = (_Float16*)take((size_t)HK * HD * SEQ * 2);
  _Float16* attO = (_Float16*)take((size_t)SEQ * DMODEL * 2);

  const int T = 256;

  // 1) precision conversion + weight transposes
  {
    int n = SEQ * DMODEL;
    cvt_f32_to_f16<<<(n + T - 1) / T, T, 0, stream>>>(x, xb, n);
  }
  transcvt<<<(DMODEL * DMODEL + T - 1) / T, T, 0, stream>>>(wq, wqT, DMODEL, DMODEL);
  transcvt<<<(DMODEL * HK * HD + T - 1) / T, T, 0, stream>>>(wk, wkT, DMODEL, HK * HD);
  transcvt<<<(DMODEL * HK * HD + T - 1) / T, T, 0, stream>>>(wv, wvT, DMODEL, HK * HD);
  transcvt<<<(DMODEL * DMODEL + T - 1) / T, T, 0, stream>>>(wo, woT, DMODEL, DMODEL);

  // 2) QKV projections (async-LDS WMMA GEMM, K=4096 compile-time)
  gemm_wmma<DMODEL><<<(SEQ / 128) * (DMODEL / 128), T, SMEM_TOTAL, stream>>>(
      xb, wqT, Qraw, SEQ, DMODEL);
  gemm_wmma<DMODEL><<<(SEQ / 128) * (HK * HD / 128), T, SMEM_TOTAL, stream>>>(
      xb, wkT, Kraw, SEQ, HK * HD);
  gemm_wmma<DMODEL><<<(SEQ / 128) * (HK * HD / 128), T, SMEM_TOTAL, stream>>>(
      xb, wvT, Vraw, SEQ, HK * HD);

  // 3) RoPE + repack to per-head f16 layouts (1/sqrt(HD) folded into Q)
  const float qscale = 0.08838834764831845f;  // 1/sqrt(128)
  rope_pack<<<(HQ * SEQ * 64 + T - 1) / T, T, 0, stream>>>(Qraw, fc, fs, Qc, HQ, qscale);
  rope_pack<<<(HK * SEQ * 64 + T - 1) / T, T, 0, stream>>>(Kraw, fc, fs, Kc, HK, 1.0f);
  transpose_v<<<(SEQ * HK * HD + T - 1) / T, T, 0, stream>>>(Vraw, Vt);

  // 4) causal GQA flash attention (WMMA)
  attn_fwd<<<(HQ * (SEQ / 16)) / 8, T, 0, stream>>>(Qc, Kc, Vt, attO);

  // 5) output projection (WMMA) -> fp32 result
  gemm_wmma<DMODEL><<<(SEQ / 128) * (DMODEL / 128), T, SMEM_TOTAL, stream>>>(
      attO, woT, out, SEQ, DMODEL);
}
